// Attention_58695023067413
// MI455X (gfx1250) — matrix operand order repaired and checked by reading, not compiled, not run on hardware
//
#include <hip/hip_runtime.h>
#include <hip/hip_bf16.h>

// ---------------------------------------------------------------------------
// Transformer block for MI455X (gfx1250, wave32, WMMA).
// All GEMMs run on v_wmma_f32_16x16x32_bf16 (bf16 operands, fp32 accum).
// Softmax quirk: normalization over the QUERY axis (column stats), /sqrt(D)
// applied after softmax -> two-pass column-stat attention, scores never hit HBM.
// This revision: __builtin_amdgcn_sched_barrier(0) pins the fragment-load
// group ahead of the WMMA group so DS loads overlap the WMMA chain (partial
// dscnt waits) instead of a full s_wait_dscnt 0 before every WMMA.
// ---------------------------------------------------------------------------

#define B_ 4
#define S_ 2048
#define E_ 1024
#define H_ 16
#define D_ 64
#define F_ 4096
#define ROWS_ (B_ * S_) // 8192

#if defined(__has_builtin) && __has_builtin(__builtin_amdgcn_sched_barrier)
#define SCHED_FENCE() __builtin_amdgcn_sched_barrier(0)
#else
#define SCHED_FENCE() __builtin_amdgcn_wave_barrier()
#endif

typedef __attribute__((ext_vector_type(16))) __bf16 v16bf;
typedef __attribute__((ext_vector_type(8)))  __bf16 v8bf;
typedef __attribute__((ext_vector_type(4)))  __bf16 v4bf;
typedef __attribute__((ext_vector_type(8)))  float  v8f;

__device__ __forceinline__ v8f zero8() {
  v8f z;
#pragma unroll
  for (int i = 0; i < 8; ++i) z[i] = 0.0f;
  return z;
}

__device__ __forceinline__ v8f wmma_bf16(v16bf a, v16bf b, v8f c) {
  // (neg_a, A, neg_b, B, c_mod, C, reuse_a, reuse_b)
  return __builtin_amdgcn_wmma_f32_16x16x32_bf16(false, a, false, b, (short)0, c,
                                                 false, false);
}

// A-fragment (16x32 bf16, MxK). Lane L: M = L&15; lanes 0-15 hold K=0..7,16..23
// at v16 elements 0..7 / 8..15; lanes 16-31 hold K=8..15,24..31. LDS row-major.
__device__ __forceinline__ v16bf load_a_frag(const __bf16* lds, int row0, int k0,
                                             int lda) {
  const int lane = threadIdx.x & 31;
  const int m = lane & 15, hi = lane >> 4;
  const __bf16* p = lds + (size_t)(row0 + m) * lda + k0 + hi * 8;
  v8bf lo = *reinterpret_cast<const v8bf*>(p);
  v8bf hg = *reinterpret_cast<const v8bf*>(p + 16);
  v16bf a;
#pragma unroll
  for (int i = 0; i < 8; ++i) { a[i] = lo[i]; a[8 + i] = hg[i]; }
  return a;
}

// B-fragment (32x16 bf16, KxN). Lane L: N = L&15; lanes 0-15 hold K=0..7,16..23,
// lanes 16-31 hold K=8..15,24..31. LDS stored [N][K] row-major (ldb = K stride).
__device__ __forceinline__ v16bf load_b_frag(const __bf16* lds, int n0, int k0,
                                             int ldb) {
  const int lane = threadIdx.x & 31;
  const int n = lane & 15, hi = lane >> 4;
  const __bf16* p = lds + (size_t)(n0 + n) * ldb + k0 + hi * 8;
  v8bf x0 = *reinterpret_cast<const v8bf*>(p);
  v8bf x1 = *reinterpret_cast<const v8bf*>(p + 16);
  v16bf b;
#pragma unroll
  for (int i = 0; i < 8; ++i) { b[i] = x0[i]; b[8 + i] = x1[i]; }
  return b;
}

// Stage fp32 global tile -> bf16 LDS (row-major), float4 reads / 8B writes.
template <int R, int C>
__device__ __forceinline__ void stage_f32_bf16(__bf16* lds, const float* g,
                                               int gld) {
  constexpr int CV = C / 4;
  for (int i = threadIdx.x; i < R * CV; i += 256) {
    int r = i / CV, c = (i % CV) * 4;
    float4 f = *reinterpret_cast<const float4*>(g + (size_t)r * gld + c);
    v4bf o;
    o[0] = (__bf16)f.x; o[1] = (__bf16)f.y; o[2] = (__bf16)f.z; o[3] = (__bf16)f.w;
    *reinterpret_cast<v4bf*>(lds + (size_t)r * C + c) = o;
  }
}

// Stage bf16 global tile -> bf16 LDS (row-major), 128-bit copies.
template <int R, int C>
__device__ __forceinline__ void stage_bf16_copy(__bf16* lds, const __bf16* g,
                                                int gld) {
  constexpr int CV = C / 8;
  for (int i = threadIdx.x; i < R * CV; i += 256) {
    int r = i / CV, c = (i % CV) * 8;
    *reinterpret_cast<uint4*>(lds + (size_t)r * C + c) =
        *reinterpret_cast<const uint4*>(g + (size_t)r * gld + c);
  }
}

// Stage fp32 weight tile [K][N] -> bf16 LDS transposed [N][K]; float4 reads.
template <int K, int N>
__device__ __forceinline__ void stage_w_transposed(__bf16* lds, const float* g,
                                                   int gld) {
  constexpr int NV = N / 4;
  for (int i = threadIdx.x; i < K * NV; i += 256) {
    int k = i / NV, n = (i % NV) * 4;
    float4 f = *reinterpret_cast<const float4*>(g + (size_t)k * gld + n);
    lds[(size_t)(n + 0) * K + k] = (__bf16)f.x;
    lds[(size_t)(n + 1) * K + k] = (__bf16)f.y;
    lds[(size_t)(n + 2) * K + k] = (__bf16)f.z;
    lds[(size_t)(n + 3) * K + k] = (__bf16)f.w;
  }
}

// ---------------------------------------------------------------------------
// K1: Q/K/V projections. grid = (ROWS/128, H). Double-buffered LDS; 12 B-frags
// loaded first (sched-fenced) -> 12 back-to-back WMMAs per K-step per wave.
// ---------------------------------------------------------------------------
__global__ __launch_bounds__(256) void k_qkv(
    const float* __restrict__ X, const float* __restrict__ WQ,
    const float* __restrict__ WK, const float* __restrict__ WV,
    __bf16* __restrict__ Qb, __bf16* __restrict__ Kb, __bf16* __restrict__ Vb) {
  __shared__ __align__(16) __bf16 sA[2][128 * 32];
  __shared__ __align__(16) __bf16 sQ[2][64 * 32];
  __shared__ __align__(16) __bf16 sK[2][64 * 32];
  __shared__ __align__(16) __bf16 sV[2][64 * 32];
  const int h = blockIdx.y;
  const int row0 = blockIdx.x * 128;
  const int wave = threadIdx.x >> 5, lane = threadIdx.x & 31;
  const int nn = lane & 15, hi = lane >> 4;
  v8f aq[4], ak[4], av[4];
#pragma unroll
  for (int j = 0; j < 4; ++j) { aq[j] = zero8(); ak[j] = zero8(); av[j] = zero8(); }

  stage_f32_bf16<128, 32>(sA[0], X + (size_t)row0 * E_, E_);
  stage_w_transposed<32, 64>(sQ[0], WQ + (size_t)h * E_ * D_, D_);
  stage_w_transposed<32, 64>(sK[0], WK + (size_t)h * E_ * D_, D_);
  stage_w_transposed<32, 64>(sV[0], WV + (size_t)h * E_ * D_, D_);
  __syncthreads();

  int buf = 0;
  for (int e0 = 0; e0 < E_; e0 += 32, buf ^= 1) {
    if (e0 + 32 < E_) {
      stage_f32_bf16<128, 32>(sA[buf ^ 1], X + (size_t)row0 * E_ + e0 + 32, E_);
      stage_w_transposed<32, 64>(sQ[buf ^ 1], WQ + ((size_t)h * E_ + e0 + 32) * D_, D_);
      stage_w_transposed<32, 64>(sK[buf ^ 1], WK + ((size_t)h * E_ + e0 + 32) * D_, D_);
      stage_w_transposed<32, 64>(sV[buf ^ 1], WV + ((size_t)h * E_ + e0 + 32) * D_, D_);
    }
    v16bf a = load_a_frag(sA[buf], wave * 16, 0, 32);
    v16bf fq[4], fk[4], fv[4];
#pragma unroll
    for (int j = 0; j < 4; ++j) {
      fq[j] = load_b_frag(sQ[buf], j * 16, 0, 32);
      fk[j] = load_b_frag(sK[buf], j * 16, 0, 32);
      fv[j] = load_b_frag(sV[buf], j * 16, 0, 32);
    }
    SCHED_FENCE(); // all DS fragment loads issue before the WMMA chain
#pragma unroll
    for (int j = 0; j < 4; ++j) {
      aq[j] = wmma_bf16(a, fq[j], aq[j]);
      ak[j] = wmma_bf16(a, fk[j], ak[j]);
      av[j] = wmma_bf16(a, fv[j], av[j]);
    }
    SCHED_FENCE();
    __syncthreads();
  }
#pragma unroll
  for (int j = 0; j < 4; ++j)
#pragma unroll
    for (int r = 0; r < 8; ++r) {
      int grow = row0 + wave * 16 + r + 8 * hi;
      int b = grow >> 11, s = grow & (S_ - 1);
      size_t idx = (((size_t)b * H_ + h) * S_ + s) * D_ + j * 16 + nn;
      Qb[idx] = (__bf16)aq[j][r];
      Kb[idx] = (__bf16)ak[j][r];
      Vb[idx] = (__bf16)av[j][r];
    }
}

// ---------------------------------------------------------------------------
// K2: column softmax stats over the QUERY axis: colmax[t], colsum[t] per (b,h).
// grid = (S/64 t-tiles, B*H). Double-buffered Q tiles; 8 B-frags preloaded.
// ---------------------------------------------------------------------------
__global__ __launch_bounds__(256) void k_colstats(
    const __bf16* __restrict__ Qb, const __bf16* __restrict__ Kb,
    float* __restrict__ cmax, float* __restrict__ csum) {
  __shared__ __align__(16) __bf16 sQ[2][128 * 64];
  __shared__ __align__(16) __bf16 sK[64 * 64];
  __shared__ float sM[8 * 64];
  __shared__ float sS[8 * 64];
  const int bh = blockIdx.y;
  const int t0 = blockIdx.x * 64;
  const __bf16* Qg = Qb + (size_t)bh * S_ * D_;
  const __bf16* Kg = Kb + (size_t)bh * S_ * D_;
  const int wave = threadIdx.x >> 5, lane = threadIdx.x & 31;
  const int nn = lane & 15, hi = lane >> 4;

  stage_bf16_copy<64, 64>(sK, Kg + (size_t)t0 * D_, D_); // [t][d] == B layout
  stage_bf16_copy<128, 64>(sQ[0], Qg, D_);
  __syncthreads();

  // preload all K-side B fragments once (block-invariant across the s loop)
  v16bf fb[8];
#pragma unroll
  for (int j = 0; j < 4; ++j) {
    fb[2 * j + 0] = load_b_frag(sK, j * 16, 0, 64);
    fb[2 * j + 1] = load_b_frag(sK, j * 16, 32, 64);
  }

  float rm[4], rs[4];
#pragma unroll
  for (int j = 0; j < 4; ++j) { rm[j] = -3.0e38f; rs[j] = 0.0f; }

  int buf = 0;
  for (int s0 = 0; s0 < S_; s0 += 128, buf ^= 1) {
    if (s0 + 128 < S_)
      stage_bf16_copy<128, 64>(sQ[buf ^ 1], Qg + (size_t)(s0 + 128) * D_, D_);
    v16bf a0 = load_a_frag(sQ[buf], wave * 16, 0, 64);
    v16bf a1 = load_a_frag(sQ[buf], wave * 16, 32, 64);
    SCHED_FENCE(); // A loads ahead of the 8-WMMA chain
    v8f c[4];
#pragma unroll
    for (int j = 0; j < 4; ++j) {
      c[j] = zero8();
      c[j] = wmma_bf16(a0, fb[2 * j + 0], c[j]);
      c[j] = wmma_bf16(a1, fb[2 * j + 1], c[j]);
    }
    SCHED_FENCE();
#pragma unroll
    for (int j = 0; j < 4; ++j) {
      float tm = c[j][0];
#pragma unroll
      for (int r = 1; r < 8; ++r) tm = fmaxf(tm, c[j][r]);
      float nm = fmaxf(rm[j], tm);
      float add = 0.0f;
#pragma unroll
      for (int r = 0; r < 8; ++r) add += __expf(c[j][r] - nm);
      rs[j] = rs[j] * __expf(rm[j] - nm) + add;
      rm[j] = nm;
    }
    __syncthreads();
  }
  // merge lane-half split (rows 0-7 in lanes 0-15, rows 8-15 in lanes 16-31)
#pragma unroll
  for (int j = 0; j < 4; ++j) {
    float om = __shfl_xor(rm[j], 16, 32);
    float os = __shfl_xor(rs[j], 16, 32);
    float nm = fmaxf(rm[j], om);
    rs[j] = rs[j] * __expf(rm[j] - nm) + os * __expf(om - nm);
    rm[j] = nm;
  }
  if (hi == 0) {
#pragma unroll
    for (int j = 0; j < 4; ++j) {
      sM[wave * 64 + j * 16 + nn] = rm[j];
      sS[wave * 64 + j * 16 + nn] = rs[j];
    }
  }
  __syncthreads();
  if (threadIdx.x < 64) {
    float m = -3.0e38f;
    for (int w = 0; w < 8; ++w) m = fmaxf(m, sM[w * 64 + threadIdx.x]);
    float s = 0.0f;
    for (int w = 0; w < 8; ++w)
      s += sS[w * 64 + threadIdx.x] * __expf(sM[w * 64 + threadIdx.x] - m);
    cmax[(size_t)bh * S_ + t0 + threadIdx.x] = m;
    csum[(size_t)bh * S_ + t0 + threadIdx.x] = s;
  }
}

// ---------------------------------------------------------------------------
// K3: vec = (exp(QK^T - colmax)/(colsum*sqrt(D))) @ V, streamed over t.
// grid = (S/128 s-tiles, B*H). P tile: D-fragment -> LDS -> A-fragment.
// V fragments preload during the probability computation.
// ---------------------------------------------------------------------------
__global__ __launch_bounds__(256) void k_attv(
    const __bf16* __restrict__ Qb, const __bf16* __restrict__ Kb,
    const __bf16* __restrict__ Vb, const float* __restrict__ cmax,
    const float* __restrict__ csum, __bf16* __restrict__ AVb) {
  __shared__ __align__(16) __bf16 sQ[128 * 64];  // [s][d]
  __shared__ __align__(16) __bf16 sK[32 * 64];   // [t][d] (B layout for scores)
  __shared__ __align__(16) __bf16 sVt[64 * 32];  // [d][t] (B layout for P@V)
  __shared__ __align__(16) __bf16 sP[128 * 32];  // [s][t] probabilities (A layout)
  const int bh = blockIdx.y, bb = bh / H_, hh = bh % H_;
  const int s0 = blockIdx.x * 128;
  const __bf16* Qg = Qb + (size_t)bh * S_ * D_;
  const __bf16* Kg = Kb + (size_t)bh * S_ * D_;
  const __bf16* Vg = Vb + (size_t)bh * S_ * D_;
  const int wave = threadIdx.x >> 5, lane = threadIdx.x & 31;
  const int nn = lane & 15, hi = lane >> 4;

  stage_bf16_copy<128, 64>(sQ, Qg + (size_t)s0 * D_, D_);
  __syncthreads();
  v16bf aq0 = load_a_frag(sQ, wave * 16, 0, 64);
  v16bf aq1 = load_a_frag(sQ, wave * 16, 32, 64);

  v8f acc[4];
#pragma unroll
  for (int j = 0; j < 4; ++j) acc[j] = zero8();

  for (int t0 = 0; t0 < S_; t0 += 32) {
    __syncthreads();
    stage_bf16_copy<32, 64>(sK, Kg + (size_t)t0 * D_, D_);
    // stage V transposed: v8bf row read + 8 strided b16 LDS writes
    for (int i = threadIdx.x; i < 32 * 8; i += 256) {
      int t = i / 8, d = (i % 8) * 8;
      v8bf v = *reinterpret_cast<const v8bf*>(Vg + (size_t)(t0 + t) * D_ + d);
#pragma unroll
      for (int u = 0; u < 8; ++u) sVt[(size_t)(d + u) * 32 + t] = v[u];
    }
    __syncthreads();
    // preload K-side and V-side fragments up front
    v16bf fk[4], fv[4];
#pragma unroll
    for (int j = 0; j < 2; ++j) {
      fk[2 * j + 0] = load_b_frag(sK, j * 16, 0, 64);
      fk[2 * j + 1] = load_b_frag(sK, j * 16, 32, 64);
    }
#pragma unroll
    for (int j = 0; j < 4; ++j) fv[j] = load_b_frag(sVt, j * 16, 0, 32);
    SCHED_FENCE(); // fragment loads ahead of score WMMAs

    v8f c0 = zero8(), c1 = zero8();
    c0 = wmma_bf16(aq0, fk[0], c0);
    c0 = wmma_bf16(aq1, fk[1], c0);
    c1 = wmma_bf16(aq0, fk[2], c1);
    c1 = wmma_bf16(aq1, fk[3], c1);
    SCHED_FENCE();
#pragma unroll
    for (int j = 0; j < 2; ++j) {
      const v8f& c = j ? c1 : c0;
      int t = t0 + j * 16 + nn;
      float cm = cmax[(size_t)bh * S_ + t];
      float inv = 1.0f / (csum[(size_t)bh * S_ + t] * 8.0f); // 8 = sqrt(D)
#pragma unroll
      for (int r = 0; r < 8; ++r) {
        float p = __expf(c[r] - cm) * inv;
        sP[(size_t)(wave * 16 + r + 8 * hi) * 32 + j * 16 + nn] = (__bf16)p;
      }
    }
    __builtin_amdgcn_wave_barrier(); // same-wave LDS ops are in-order; fence sched
    v16bf ap = load_a_frag(sP, wave * 16, 0, 32);
    SCHED_FENCE();
#pragma unroll
    for (int j = 0; j < 4; ++j) acc[j] = wmma_bf16(ap, fv[j], acc[j]);
    SCHED_FENCE();
  }
  // write vec into [B,S,H*D] layout (heads interleaved for WO GEMM)
#pragma unroll
  for (int j = 0; j < 4; ++j)
#pragma unroll
    for (int r = 0; r < 8; ++r) {
      int s = s0 + wave * 16 + r + 8 * hi;
      AVb[((size_t)bb * S_ + s) * (H_ * D_) + hh * D_ + j * 16 + nn] =
          (__bf16)acc[j][r];
    }
}

// ---------------------------------------------------------------------------
// K4: Z = AV @ WO + X (fp32 out). grid = (ROWS/128, E/64). Double-buffered.
// ---------------------------------------------------------------------------
__global__ __launch_bounds__(256) void k_gemm_wo(
    const __bf16* __restrict__ A, const float* __restrict__ WO,
    const float* __restrict__ X, float* __restrict__ Z) {
  __shared__ __align__(16) __bf16 sA[2][128 * 32];
  __shared__ __align__(16) __bf16 sB[2][64 * 32];
  const int row0 = blockIdx.x * 128, col0 = blockIdx.y * 64;
  const int wave = threadIdx.x >> 5, lane = threadIdx.x & 31;
  const int nn = lane & 15, hi = lane >> 4;
  v8f acc[4];
#pragma unroll
  for (int j = 0; j < 4; ++j) acc[j] = zero8();

  stage_bf16_copy<128, 32>(sA[0], A + (size_t)row0 * E_, E_);
  stage_w_transposed<32, 64>(sB[0], WO + col0, E_);
  __syncthreads();

  int buf = 0;
  for (int k0 = 0; k0 < E_; k0 += 32, buf ^= 1) {
    if (k0 + 32 < E_) {
      stage_bf16_copy<128, 32>(sA[buf ^ 1], A + (size_t)row0 * E_ + k0 + 32, E_);
      stage_w_transposed<32, 64>(sB[buf ^ 1], WO + (size_t)(k0 + 32) * E_ + col0, E_);
    }
    v16bf a = load_a_frag(sA[buf], wave * 16, 0, 32);
    v16bf fb[4];
#pragma unroll
    for (int j = 0; j < 4; ++j) fb[j] = load_b_frag(sB[buf], j * 16, 0, 32);
    SCHED_FENCE();
#pragma unroll
    for (int j = 0; j < 4; ++j) acc[j] = wmma_bf16(a, fb[j], acc[j]);
    SCHED_FENCE();
    __syncthreads();
  }
#pragma unroll
  for (int j = 0; j < 4; ++j)
#pragma unroll
    for (int r = 0; r < 8; ++r) {
      int grow = row0 + wave * 16 + r + 8 * hi;
      int col = col0 + j * 16 + nn;
      Z[(size_t)grow * E_ + col] = acc[j][r] + X[(size_t)grow * E_ + col];
    }
}

// ---------------------------------------------------------------------------
// K5: H = relu(Y1 @ W1 + b1) (bf16 out). grid = (ROWS/128, F/64).
// ---------------------------------------------------------------------------
__global__ __launch_bounds__(256) void k_ffn1(
    const float* __restrict__ Y1, const float* __restrict__ W1,
    const float* __restrict__ b1, __bf16* __restrict__ Hm) {
  __shared__ __align__(16) __bf16 sA[2][128 * 32];
  __shared__ __align__(16) __bf16 sB[2][64 * 32];
  const int row0 = blockIdx.x * 128, col0 = blockIdx.y * 64;
  const int wave = threadIdx.x >> 5, lane = threadIdx.x & 31;
  const int nn = lane & 15, hi = lane >> 4;
  v8f acc[4];
#pragma unroll
  for (int j = 0; j < 4; ++j) acc[j] = zero8();

  stage_f32_bf16<128, 32>(sA[0], Y1 + (size_t)row0 * E_, E_);
  stage_w_transposed<32, 64>(sB[0], W1 + col0, F_);
  __syncthreads();

  int buf = 0;
  for (int k0 = 0; k0 < E_; k0 += 32, buf ^= 1) {
    if (k0 + 32 < E_) {
      stage_f32_bf16<128, 32>(sA[buf ^ 1], Y1 + (size_t)row0 * E_ + k0 + 32, E_);
      stage_w_transposed<32, 64>(sB[buf ^ 1], W1 + (size_t)(k0 + 32) * F_ + col0, F_);
    }
    v16bf a = load_a_frag(sA[buf], wave * 16, 0, 32);
    v16bf fb[4];
#pragma unroll
    for (int j = 0; j < 4; ++j) fb[j] = load_b_frag(sB[buf], j * 16, 0, 32);
    SCHED_FENCE();
#pragma unroll
    for (int j = 0; j < 4; ++j) acc[j] = wmma_bf16(a, fb[j], acc[j]);
    SCHED_FENCE();
    __syncthreads();
  }
#pragma unroll
  for (int j = 0; j < 4; ++j)
#pragma unroll
    for (int r = 0; r < 8; ++r) {
      int grow = row0 + wave * 16 + r + 8 * hi;
      int col = col0 + j * 16 + nn;
      float v = acc[j][r] + b1[col];
      Hm[(size_t)grow * F_ + col] = (__bf16)(v > 0.0f ? v : 0.0f);
    }
}

// ---------------------------------------------------------------------------
// K6: Z2 = H @ W2 + b2 + Y1 (fp32 out). grid = (ROWS/128, E/64). K = 4096.
// ---------------------------------------------------------------------------
__global__ __launch_bounds__(256) void k_ffn2(
    const __bf16* __restrict__ Hm, const float* __restrict__ W2,
    const float* __restrict__ b2, const float* __restrict__ Y1,
    float* __restrict__ Z2) {
  __shared__ __align__(16) __bf16 sA[2][128 * 32];
  __shared__ __align__(16) __bf16 sB[2][64 * 32];
  const int row0 = blockIdx.x * 128, col0 = blockIdx.y * 64;
  const int wave = threadIdx.x >> 5, lane = threadIdx.x & 31;
  const int nn = lane & 15, hi = lane >> 4;
  v8f acc[4];
#pragma unroll
  for (int j = 0; j < 4; ++j) acc[j] = zero8();

  stage_bf16_copy<128, 32>(sA[0], Hm + (size_t)row0 * F_, F_);
  stage_w_transposed<32, 64>(sB[0], W2 + col0, E_);
  __syncthreads();

  int buf = 0;
  for (int k0 = 0; k0 < F_; k0 += 32, buf ^= 1) {
    if (k0 + 32 < F_) {
      stage_bf16_copy<128, 32>(sA[buf ^ 1], Hm + (size_t)row0 * F_ + k0 + 32, F_);
      stage_w_transposed<32, 64>(sB[buf ^ 1], W2 + (size_t)(k0 + 32) * E_ + col0, E_);
      __builtin_prefetch(Hm + (size_t)row0 * F_ + k0 + 64, 0, 1);
    }
    v16bf a = load_a_frag(sA[buf], wave * 16, 0, 32);
    v16bf fb[4];
#pragma unroll
    for (int j = 0; j < 4; ++j) fb[j] = load_b_frag(sB[buf], j * 16, 0, 32);
    SCHED_FENCE();
#pragma unroll
    for (int j = 0; j < 4; ++j) acc[j] = wmma_bf16(a, fb[j], acc[j]);
    SCHED_FENCE();
    __syncthreads();
  }
#pragma unroll
  for (int j = 0; j < 4; ++j)
#pragma unroll
    for (int r = 0; r < 8; ++r) {
      int grow = row0 + wave * 16 + r + 8 * hi;
      int col = col0 + j * 16 + nn;
      Z2[(size_t)grow * E_ + col] =
          acc[j][r] + b2[col] + Y1[(size_t)grow * E_ + col];
    }
}

// ---------------------------------------------------------------------------
// LayerNorm over E=1024 per row. grid = ROWS, block = 256 (4 elems/thread).
// ---------------------------------------------------------------------------
__global__ __launch_bounds__(256) void k_layernorm(
    const float* __restrict__ Zin, const float* __restrict__ gamma,
    const float* __restrict__ beta, float* __restrict__ Yout) {
  __shared__ float wsum[8];
  __shared__ float wvar[8];
  const int row = blockIdx.x;
  const float* z = Zin + (size_t)row * E_;
  float v[4];
  float s = 0.0f;
#pragma unroll
  for (int i = 0; i < 4; ++i) { v[i] = z[threadIdx.x + i * 256]; s += v[i]; }
#pragma unroll
  for (int o = 16; o > 0; o >>= 1) s += __shfl_xor(s, o, 32);
  if ((threadIdx.x & 31) == 0) wsum[threadIdx.x >> 5] = s;
  __syncthreads();
  float mean = 0.0f;
  for (int w = 0; w < 8; ++w) mean += wsum[w];
  mean *= (1.0f / E_);
  float q = 0.0f;
#pragma unroll
  for (int i = 0; i < 4; ++i) { float d = v[i] - mean; q += d * d; }
#pragma unroll
  for (int o = 16; o > 0; o >>= 1) q += __shfl_xor(q, o, 32);
  if ((threadIdx.x & 31) == 0) wvar[threadIdx.x >> 5] = q;
  __syncthreads();
  float var = 0.0f;
  for (int w = 0; w < 8; ++w) var += wvar[w];
  var *= (1.0f / E_);
  float rstd = rsqrtf(var + 1e-5f);
#pragma unroll
  for (int i = 0; i < 4; ++i) {
    int c = threadIdx.x + i * 256;
    Yout[(size_t)row * E_ + c] = (v[i] - mean) * rstd * gamma[c] + beta[c];
  }
}

// ---------------------------------------------------------------------------
// Workspace layout (bytes). Regions reused once producers are dead:
//   phase A: Qb|Kb|Vb|cmax|csum|AVb  (0 .. 65 MiB), Z (65..97), Y1 (97..129)
//   phase B: Hmid overlays 0..64 MiB (Q/K/V/stats/AV dead), Z2 overlays Z.
// Total required: ~129 MiB.
// ---------------------------------------------------------------------------
static constexpr size_t SZ_QKV = (size_t)B_ * H_ * S_ * D_ * 2;  // 16 MiB
static constexpr size_t SZ_STAT = (size_t)B_ * H_ * S_ * 4;      // 512 KiB
static constexpr size_t SZ_AV = (size_t)ROWS_ * E_ * 2;          // 16 MiB
static constexpr size_t SZ_Z = (size_t)ROWS_ * E_ * 4;           // 32 MiB
static constexpr size_t OFF_QB = 0;
static constexpr size_t OFF_KB = OFF_QB + SZ_QKV;
static constexpr size_t OFF_VB = OFF_KB + SZ_QKV;
static constexpr size_t OFF_CM = OFF_VB + SZ_QKV;
static constexpr size_t OFF_CS = OFF_CM + SZ_STAT;
static constexpr size_t OFF_AV = OFF_CS + SZ_STAT;
static constexpr size_t OFF_Z = OFF_AV + SZ_AV;
static constexpr size_t OFF_Y1 = OFF_Z + SZ_Z;
static constexpr size_t OFF_HM = 0;     // overlays Q/K/V/stats/AV (dead)
static constexpr size_t OFF_Z2 = OFF_Z; // overlays Z (dead after LN1)

extern "C" void kernel_launch(void* const* d_in, const int* in_sizes, int n_in,
                              void* d_out, int out_size, void* d_ws,
                              size_t ws_size, hipStream_t stream) {
  (void)in_sizes; (void)n_in; (void)out_size; (void)ws_size;
  const float* X = (const float*)d_in[0];
  const float* WQ = (const float*)d_in[1];
  const float* WK = (const float*)d_in[2];
  const float* WV = (const float*)d_in[3];
  const float* WO = (const float*)d_in[4];
  const float* gamma = (const float*)d_in[5];
  const float* beta = (const float*)d_in[6];
  const float* W1 = (const float*)d_in[7];
  const float* b1 = (const float*)d_in[8];
  const float* W2 = (const float*)d_in[9];
  const float* b2 = (const float*)d_in[10];

  char* ws = (char*)d_ws;
  __bf16* Qb = (__bf16*)(ws + OFF_QB);
  __bf16* Kb = (__bf16*)(ws + OFF_KB);
  __bf16* Vb = (__bf16*)(ws + OFF_VB);
  float* cmax = (float*)(ws + OFF_CM);
  float* csum = (float*)(ws + OFF_CS);
  __bf16* AVb = (__bf16*)(ws + OFF_AV);
  float* Z = (float*)(ws + OFF_Z);
  float* Y1 = (float*)(ws + OFF_Y1);
  __bf16* Hm = (__bf16*)(ws + OFF_HM);
  float* Z2 = (float*)(ws + OFF_Z2);
  float* out = (float*)d_out;

  dim3 blk(256);
  k_qkv<<<dim3(ROWS_ / 128, H_), blk, 0, stream>>>(X, WQ, WK, WV, Qb, Kb, Vb);
  k_colstats<<<dim3(S_ / 64, B_ * H_), blk, 0, stream>>>(Qb, Kb, cmax, csum);
  k_attv<<<dim3(S_ / 128, B_ * H_), blk, 0, stream>>>(Qb, Kb, Vb, cmax, csum,
                                                      AVb);
  k_gemm_wo<<<dim3(ROWS_ / 128, E_ / 64), blk, 0, stream>>>(AVb, WO, X, Z);
  k_layernorm<<<dim3(ROWS_), blk, 0, stream>>>(Z, gamma, beta, Y1);
  k_ffn1<<<dim3(ROWS_ / 128, F_ / 64), blk, 0, stream>>>(Y1, W1, b1, Hm);
  k_ffn2<<<dim3(ROWS_ / 128, E_ / 64), blk, 0, stream>>>(Hm, W2, b2, Y1, Z2);
  k_layernorm<<<dim3(ROWS_), blk, 0, stream>>>(Z2, gamma, beta, out);
}
